// MultiheadSelfAttention_29076928594353
// MI455X (gfx1250) — matrix operand order repaired and checked by reading, not compiled, not run on hardware
//
#include <hip/hip_runtime.h>

// ---------------------------------------------------------------------------
// MI455X (gfx1250) multi-head self-attention with interaction bias.
// bf16 WMMA (v_wmma_f32_16x16x32_bf16) for all matmuls, f32 softmax.
// LDS-staged shared operands; ping-pong (unroll-2) pipelined GEMM fragments.
// ---------------------------------------------------------------------------

typedef __bf16 bf16;
typedef __attribute__((ext_vector_type(8)))  __bf16 v8bf;
typedef __attribute__((ext_vector_type(16))) __bf16 v16bf;
typedef __attribute__((ext_vector_type(8)))  float  v8f;

#define DIM   512
#define HEADS 8
#define HD    64
#define BB    32
#define NN    512
#define MTOT  (BB * NN)          // 16384 rows
#define C3    (3 * DIM)          // 1536

// LDS row paddings (bf16 elements) chosen for conflict-free b128 lane access
#define BPAD  8                  // Wt / V rows: 512+8  -> dword stride 260 (4n banks)
#define KPAD  8                  // K rows:      64+8   -> dword stride 36  (distinct banks)
#define PPAD  4                  // P rows (f32): 32+4  -> stride 36 dwords

__device__ __forceinline__ v8f wzero() {
  v8f z = {0.f, 0.f, 0.f, 0.f, 0.f, 0.f, 0.f, 0.f};
  return z;
}

// A-matrix fragment, 16x32 bf16, row-major source with leading dim `ld`.
// Lanes 0-15: row M=lane, K 0-7 & 16-23; lanes 16-31: same rows, K 8-15 & 24-31.
__device__ __forceinline__ v16bf frag_a(const bf16* __restrict__ base, int ld) {
  int lane = threadIdx.x & 31;
  const bf16* p = base + (size_t)(lane & 15) * ld + (lane >> 4) * 8;
  v8bf lo = *(const v8bf*)(p);
  v8bf hi = *(const v8bf*)(p + 16);
  return __builtin_shufflevector(lo, hi, 0,1,2,3,4,5,6,7,8,9,10,11,12,13,14,15);
}

// B-matrix fragment, 32x16 bf16, from (N x K)-major source: B[kk][n] = src[n*ld + kk].
// Lanes 0-15: col N=lane, K 0-7 & 16-23; lanes 16-31: col N=lane-16, K 8-15 & 24-31.
__device__ __forceinline__ v16bf frag_bt(const bf16* __restrict__ base, int ld) {
  int lane = threadIdx.x & 31;
  const bf16* p = base + (size_t)(lane & 15) * ld + (lane >> 4) * 8;
  v8bf lo = *(const v8bf*)(p);
  v8bf hi = *(const v8bf*)(p + 16);
  return __builtin_shufflevector(lo, hi, 0,1,2,3,4,5,6,7,8,9,10,11,12,13,14,15);
}

__device__ __forceinline__ v8f wmma_bf(v16bf a, v16bf b, v8f c) {
  return __builtin_amdgcn_wmma_f32_16x16x32_bf16(false, a, false, b,
                                                 (short)0, c, false, false);
}

// ---------------------------------------------------------------------------
// Stage 1a: f32 -> bf16 copy
__global__ __launch_bounds__(256) void k_cvt_bf16(bf16* __restrict__ dst,
                                                  const float* __restrict__ src,
                                                  int n) {
  int i = blockIdx.x * 256 + threadIdx.x;
  if (i < n) dst[i] = (bf16)src[i];
}

// Stage 1b: transpose-convert weight (K x Cout) f32 -> (Cout x K) bf16
__global__ __launch_bounds__(256) void k_cvt_t_bf16(bf16* __restrict__ dst,
                                                    const float* __restrict__ src,
                                                    int K, int Cout) {
  int i = blockIdx.x * 256 + threadIdx.x;
  if (i >= K * Cout) return;
  int c = i / K;
  int kk = i - c * K;
  dst[i] = (bf16)src[(size_t)kk * Cout + c];
}

// ---------------------------------------------------------------------------
// bf16 WMMA GEMM (K = DIM = 512): out[m,c] = sum_k A[m,k]*Wt[c,k] + bias[c]
// Block = 256 threads (8 waves); tile = 128 rows x 64 cols; the 64x512 Wt
// panel (shared by all 8 waves) staged once in LDS.  K-loop is ping-pong
// pipelined (step 64, two independent fragment sets): loads land directly in
// their final registers (no rotation movs) and each WMMA quad executes with
// the other stage's loads in flight.
// mode 0: scatter into Q (B,H,N,hd), K (B,H,N,hd), Vt (B,H,hd,N) as bf16
// mode 1: dense f32 output (M x Cout)
__global__ __launch_bounds__(256) void k_gemm(const bf16* __restrict__ A,
                                              const bf16* __restrict__ Wt,
                                              const float* __restrict__ bias,
                                              int Cout, int mode,
                                              bf16* __restrict__ Qo,
                                              bf16* __restrict__ Ko,
                                              bf16* __restrict__ Vto,
                                              float* __restrict__ Fo) {
  extern __shared__ char dynsmem[];
  bf16* Bsh = (bf16*)dynsmem;                     // [64][DIM + BPAD]
  const int BLD = DIM + BPAD;                     // 520

  int tid  = threadIdx.x;
  int wave = tid >> 5;
  int lane = tid & 31;
  int sel  = lane >> 4;
  int nn   = lane & 15;

  int cbase = blockIdx.x * 64;
  int mbase = blockIdx.y * 128 + wave * 16;

  // ---- cooperative stage of the 64 x 512 Wt panel (4096 16B chunks) ------
  {
    const bf16* Wp = Wt + (size_t)cbase * DIM;
#pragma unroll
    for (int i = 0; i < 16; ++i) {
      int idx = i * 256 + tid;                    // 0..4095
      int row = idx >> 6;                         // 64 chunks per row
      int c   = idx & 63;
      v8bf v  = *(const v8bf*)(Wp + (size_t)row * DIM + c * 8);
      *(v8bf*)(Bsh + row * BLD + c * 8) = v;
    }
  }
  __syncthreads();

  v8f acc[4] = {wzero(), wzero(), wzero(), wzero()};

  const bf16* Arow = A + (size_t)mbase * DIM;

  // stage-A fragments for kb = 0
  v16bf afA = frag_a(Arow, DIM);
  v16bf bfA[4];
#pragma unroll
  for (int t = 0; t < 4; ++t) bfA[t] = frag_bt(Bsh + (t * 16) * BLD, BLD);

  for (int kb = 0; kb < DIM; kb += 64) {
    // load stage-B fragments (kb+32) while stage-A computes
    __builtin_prefetch(Arow + kb + 64, 0, 1);
    v16bf afB = frag_a(Arow + kb + 32, DIM);
    v16bf bfB[4];
#pragma unroll
    for (int t = 0; t < 4; ++t)
      bfB[t] = frag_bt(Bsh + (t * 16) * BLD + kb + 32, BLD);

#pragma unroll
    for (int t = 0; t < 4; ++t) acc[t] = wmma_bf(afA, bfA[t], acc[t]);

    // load stage-A fragments for kb+64 (wraps harmlessly on last iteration)
    int kn = (kb + 64 < DIM) ? kb + 64 : 0;
    afA = frag_a(Arow + kn, DIM);
#pragma unroll
    for (int t = 0; t < 4; ++t)
      bfA[t] = frag_bt(Bsh + (t * 16) * BLD + kn, BLD);

#pragma unroll
    for (int t = 0; t < 4; ++t) acc[t] = wmma_bf(afB, bfB[t], acc[t]);
  }

#pragma unroll
  for (int t = 0; t < 4; ++t) {
#pragma unroll
    for (int r = 0; r < 8; ++r) {
      int row = mbase + r + 8 * sel;
      int col = cbase + t * 16 + nn;
      float val = acc[t][r] + bias[col];
      if (mode == 0) {
        int b = row >> 9, n = row & 511;
        int which = col >> 9, rem = col & 511;
        int h = rem >> 6, d = rem & 63;
        size_t bh = (size_t)(b * HEADS + h);
        if (which == 0)      Qo[(bh * NN + n) * HD + d] = (bf16)val;
        else if (which == 1) Ko[(bh * NN + n) * HD + d] = (bf16)val;
        else                 Vto[(bh * HD + d) * NN + n] = (bf16)val;  // transposed
      } else {
        Fo[(size_t)row * Cout + col] = val;
      }
    }
  }
}

// ---------------------------------------------------------------------------
// Stage 3: flash attention.  One 1024-thread block (32 waves) per (b,h);
// wave w owns query tile w.  Whole K (512x64) and Vt (64x512) panels staged
// once in LDS (shared by all 32 waves).  Interaction/mask gathers (the HBM
// roofline stream) are issued at the top of each 32-key step so they overlap
// the score WMMAs.
__global__ __launch_bounds__(1024) void k_attn(const bf16* __restrict__ Q,
                                               const bf16* __restrict__ Kb,
                                               const bf16* __restrict__ Vt,
                                               const float* __restrict__ inter,
                                               const unsigned char* __restrict__ maskp,
                                               bf16* __restrict__ Ao) {
  extern __shared__ char dynsmem[];
  const int KLD = HD + KPAD;                      // 72
  const int VLD = NN + BPAD;                      // 520
  const int PLD = 32 + PPAD;                      // 36 floats
  bf16*  Ksh = (bf16*)dynsmem;                    // [NN][KLD]
  bf16*  Vsh = Ksh + NN * KLD;                    // [HD][VLD]
  float* Psh = (float*)(Vsh + HD * VLD);          // [32 waves][16*PLD]

  int tid  = threadIdx.x;
  int wave = tid >> 5;
  int lane = tid & 31;
  int sel  = lane >> 4;
  int nn   = lane & 15;

  int bh = blockIdx.x;                            // b*H + h
  int b  = bh >> 3;
  int h  = bh & 7;
  int qbase = wave * 16;                          // wave == query tile

  // ---- cooperative stage of K and Vt panels ------------------------------
  {
    const bf16* Kp = Kb + (size_t)bh * NN * HD;
    const bf16* Vp = Vt + (size_t)bh * HD * NN;
#pragma unroll
    for (int i = 0; i < 4; ++i) {                 // K: 4096 chunks of 8 bf16
      int idx = i * 1024 + tid;
      int row = idx >> 3;                         // 8 chunks per 64-elem row
      int c   = idx & 7;
      v8bf v  = *(const v8bf*)(Kp + (size_t)row * HD + c * 8);
      *(v8bf*)(Ksh + row * KLD + c * 8) = v;
    }
#pragma unroll
    for (int i = 0; i < 4; ++i) {                 // Vt: 4096 chunks of 8 bf16
      int idx = i * 1024 + tid;
      int row = idx >> 6;                         // 64 chunks per 512-elem row
      int c   = idx & 63;
      v8bf v  = *(const v8bf*)(Vp + (size_t)row * NN + c * 8);
      *(v8bf*)(Vsh + row * VLD + c * 8) = v;
    }
  }
  __syncthreads();

  const bf16* Qbase = Q + ((size_t)bh * NN + qbase) * HD;
  v16bf qf0 = frag_a(Qbase, HD);                  // hd 0..31
  v16bf qf1 = frag_a(Qbase + 32, HD);             // hd 32..63

  v8f o[4] = {wzero(), wzero(), wzero(), wzero()};
  float mloc[8], lloc[8];
#pragma unroll
  for (int r = 0; r < 8; ++r) { mloc[r] = -3.0e38f; lloc[r] = 0.f; }

  float* ps = Psh + wave * 16 * PLD;

  for (int kb = 0; kb < NN; kb += 32) {
    // ---- issue interaction + mask gathers first (overlap with WMMAs) -----
    int key0 = kb + nn;
    int key1 = kb + 16 + nn;
    float bias0[8], bias1[8];
#pragma unroll
    for (int r = 0; r < 8; ++r) {
      int row = qbase + r + 8 * sel;
      size_t rbase = ((size_t)b * NN + row) * NN;
      bias0[r] = inter[(rbase + key0) * HEADS + h] +
                 (maskp[rbase + key0] ? 0.f : -1e9f);
      bias1[r] = inter[(rbase + key1) * HEADS + h] +
                 (maskp[rbase + key1] ? 0.f : -1e9f);
    }
    {
      int kbn = (kb + 32 < NN) ? kb + 32 : kb;    // prefetch next HBM block
      const float* pf =
          &inter[((((size_t)b * NN + qbase + sel * 8) * NN) + kbn + nn) * HEADS + h];
      __builtin_prefetch(pf, 0, 1);
    }

    // ---- scores for 32 keys: batch all K-fragments, then WMMA quad -------
    v16bf kf[4];
    kf[0] = frag_bt(Ksh + (kb) * KLD, KLD);
    kf[1] = frag_bt(Ksh + (kb) * KLD + 32, KLD);
    kf[2] = frag_bt(Ksh + (kb + 16) * KLD, KLD);
    kf[3] = frag_bt(Ksh + (kb + 16) * KLD + 32, KLD);
    v8f s0 = wzero(), s1 = wzero();
    s0 = wmma_bf(qf0, kf[0], s0);
    s0 = wmma_bf(qf1, kf[1], s0);
    s1 = wmma_bf(qf0, kf[2], s1);
    s1 = wmma_bf(qf1, kf[3], s1);

#pragma unroll
    for (int r = 0; r < 8; ++r) {
      s0[r] = s0[r] * 0.125f + bias0[r];
      s1[r] = s1[r] * 0.125f + bias1[r];
    }

    // ---- online softmax update ------------------------------------------
    float alpha[8];
#pragma unroll
    for (int r = 0; r < 8; ++r) {
      float x = fmaxf(s0[r], s1[r]);
      x = fmaxf(x, __shfl_xor(x, 1, 32));
      x = fmaxf(x, __shfl_xor(x, 2, 32));
      x = fmaxf(x, __shfl_xor(x, 4, 32));
      x = fmaxf(x, __shfl_xor(x, 8, 32));         // row max within 16-lane half
      float nm = fmaxf(mloc[r], x);
      alpha[r] = __expf(mloc[r] - nm);
      float p0 = __expf(s0[r] - nm);
      float p1 = __expf(s1[r] - nm);
      float rs = p0 + p1;
      rs += __shfl_xor(rs, 1, 32);
      rs += __shfl_xor(rs, 2, 32);
      rs += __shfl_xor(rs, 4, 32);
      rs += __shfl_xor(rs, 8, 32);
      lloc[r] = lloc[r] * alpha[r] + rs;
      mloc[r] = nm;
      s0[r] = p0;
      s1[r] = p1;
    }
#pragma unroll
    for (int t = 0; t < 4; ++t)
#pragma unroll
      for (int r = 0; r < 8; ++r) o[t][r] *= alpha[r];

    // ---- P: C-layout -> A-layout via per-wave LDS (same-wave DS in-order)
#pragma unroll
    for (int r = 0; r < 8; ++r) {
      int rr = r + 8 * sel;
      ps[rr * PLD + nn]      = s0[r];
      ps[rr * PLD + 16 + nn] = s1[r];
    }
    const float* pr = ps + (lane & 15) * PLD + sel * 8;
    v16bf pfrag;
#pragma unroll
    for (int e = 0; e < 8; ++e) {
      pfrag[e]     = (bf16)pr[e];
      pfrag[8 + e] = (bf16)pr[16 + e];
    }

    // ---- O += P @ V: batch all V-fragments, then WMMA quad ---------------
    v16bf vf[4];
#pragma unroll
    for (int t = 0; t < 4; ++t)
      vf[t] = frag_bt(Vsh + (t * 16) * VLD + kb, VLD);
#pragma unroll
    for (int t = 0; t < 4; ++t)
      o[t] = wmma_bf(pfrag, vf[t], o[t]);
  }

  // ---- normalize and write (B, N, H*hd) bf16 -----------------------------
#pragma unroll
  for (int r = 0; r < 8; ++r) {
    float inv = 1.f / lloc[r];
    int row = qbase + r + 8 * sel;
    size_t obase = ((size_t)b * NN + row) * DIM + h * HD;
#pragma unroll
    for (int t = 0; t < 4; ++t)
      Ao[obase + t * 16 + nn] = (bf16)(o[t][r] * inv);
  }
}

// ---------------------------------------------------------------------------
extern "C" void kernel_launch(void* const* d_in, const int* in_sizes, int n_in,
                              void* d_out, int out_size, void* d_ws, size_t ws_size,
                              hipStream_t stream) {
  const float*         inputs = (const float*)d_in[0];
  const unsigned char* mask   = (const unsigned char*)d_in[1];
  const float*         inter  = (const float*)d_in[2];
  const float*         Wqkv   = (const float*)d_in[3];
  const float*         bqkv   = (const float*)d_in[4];
  const float*         Wout   = (const float*)d_in[5];
  const float*         bout   = (const float*)d_in[6];
  float*               out    = (float*)d_out;

  // workspace carve-up (bf16, all regions 128B+ aligned)
  bf16* w = (bf16*)d_ws;
  bf16* Xb     = w;                 w += (size_t)MTOT * DIM;     // 16 MB
  bf16* WqkvT  = w;                 w += (size_t)C3 * DIM;       // 1.5 MB
  bf16* WoutT  = w;                 w += (size_t)DIM * DIM;      // 0.5 MB
  bf16* Qb     = w;                 w += (size_t)BB * HEADS * NN * HD;
  bf16* Kb     = w;                 w += (size_t)BB * HEADS * NN * HD;
  bf16* Vt     = w;                 w += (size_t)BB * HEADS * HD * NN;
  bf16* Aout   = w;                 w += (size_t)MTOT * DIM;

  // Stage 1: precision/layout conversion
  {
    int nx = MTOT * DIM;
    k_cvt_bf16<<<nx / 256, 256, 0, stream>>>(Xb, inputs, nx);
    int nq = DIM * C3;
    k_cvt_t_bf16<<<nq / 256, 256, 0, stream>>>(WqkvT, Wqkv, DIM, C3);
    int no = DIM * DIM;
    k_cvt_t_bf16<<<no / 256, 256, 0, stream>>>(WoutT, Wout, DIM, DIM);
  }

  size_t gemm_lds = (size_t)64 * (DIM + BPAD) * sizeof(bf16);            // 66,560 B
  size_t attn_lds = ((size_t)NN * (HD + KPAD) + (size_t)HD * (NN + BPAD)) * sizeof(bf16)
                  + (size_t)32 * 16 * (32 + PPAD) * sizeof(float);       // ~214 KB

  // Stage 2: fused QKV projection (M=16384, K=512, Cout=1536), scatter Q/K/Vt
  k_gemm<<<dim3(C3 / 64, MTOT / 128), 256, gemm_lds, stream>>>(
      Xb, WqkvT, bqkv, C3, /*mode=*/0, Qb, Kb, Vt, nullptr);

  // Stage 3: attention (one block per (b,h), 32 waves = 32 query tiles)
  k_attn<<<BB * HEADS, 1024, attn_lds, stream>>>(Qb, Kb, Vt, inter, mask, Aout);

  // Stage 4: output projection (M=16384, K=512, Cout=512) -> f32 out + bias
  k_gemm<<<dim3(DIM / 64, MTOT / 128), 256, gemm_lds, stream>>>(
      Aout, WoutT, bout, DIM, /*mode=*/1, nullptr, nullptr, nullptr, out);
}
